// Attention_73615739453902
// MI455X (gfx1250) — matrix operand order repaired and checked by reading, not compiled, not run on hardware
//
#include <hip/hip_runtime.h>
#include <math.h>

// Problem constants (from reference setup_inputs)
#define B_ 32
#define S_ 2048
#define H_ 512
#define C_CLIP 10.0f

typedef __attribute__((ext_vector_type(16))) __bf16        v16bf;
typedef __attribute__((ext_vector_type(8)))  float         v8f;
typedef __attribute__((ext_vector_type(4)))  unsigned int  v4u;
typedef __attribute__((ext_vector_type(2)))  unsigned int  v2u;
typedef __attribute__((ext_vector_type(4)))  unsigned int  u32x4;
typedef __attribute__((ext_vector_type(8)))  int           i32x8;

__device__ __forceinline__ unsigned short f2bf(float f) {
  union { float f; unsigned u; } x; x.f = f;
  unsigned r = x.u + 0x7FFFu + ((x.u >> 16) & 1u);   // round-to-nearest-even
  return (unsigned short)(r >> 16);
}

// tanh via exp: ~5 VALU ops (v_exp_f32 + rcp) instead of ~20 for ocml tanhf.
__device__ __forceinline__ float fast_tanh(float x) {
  float cx = fminf(fmaxf(x, -15.0f), 15.0f);
  float e = __expf(2.0f * cx);
  return (e - 1.0f) * __builtin_amdgcn_rcpf(e + 1.0f);
}

// Issue a 2D TDM load: 512 rows x 64B (row stride 1024B global), LDS rows
// padded to 80B via pad_interval=16 DWORDs / pad_amount=4 DWORDs.
__device__ __forceinline__ void tdm_issue_b(unsigned lds_addr,
                                            unsigned long long ga,
                                            i32x8 g1) {
  u32x4 g0;
  g0.x = 1u;                                               // count = 1
  g0.y = lds_addr;                                         // lds_addr
  g0.z = (unsigned)ga;                                     // global_addr lo
  g0.w = (unsigned)((ga >> 32) & 0x1FFFFFFull) | (2u << 30);  // hi | type=2
  asm volatile("tensor_load_to_lds %0, %1" :: "s"(g0), "s"(g1) : "memory");
}

// ---------------------------------------------------------------------------
// Kernel 0: pre-convert W_ref (fp32 [o][h]) -> bf16 once ([n][k] layout).
// ---------------------------------------------------------------------------
__global__ __launch_bounds__(256) void wconv_kernel(const float* __restrict__ W,
                                                    unsigned short* __restrict__ Wbf) {
  int i4 = (blockIdx.x * 256 + threadIdx.x) * 4;
  float4 w4 = *(const float4*)(W + i4);
  v2u p;
  p.x = (unsigned)f2bf(w4.x) | ((unsigned)f2bf(w4.y) << 16);
  p.y = (unsigned)f2bf(w4.z) | ((unsigned)f2bf(w4.w) << 16);
  *(v2u*)(Wbf + i4) = p;
}

// ---------------------------------------------------------------------------
// Kernel 1: q[b,o] = sum_h tgt[b,h] * W_q[o,h]
// ---------------------------------------------------------------------------
__global__ __launch_bounds__(256) void q_kernel(const float* __restrict__ tgt,
                                                const float* __restrict__ Wq,
                                                float* __restrict__ qout) {
  int gid = blockIdx.x * 256 + threadIdx.x;
  int b = gid >> 9;
  int o = gid & (H_ - 1);
  const float* t = tgt + (size_t)b * H_;
  const float* w = Wq + (size_t)o * H_;
  float acc = 0.0f;
  for (int h = 0; h < H_; h += 4) {
    float4 tv = *(const float4*)(t + h);
    float4 wv = *(const float4*)(w + h);
    acc += tv.x * wv.x + tv.y * wv.y + tv.z * wv.z + tv.w * wv.w;
  }
  qout[gid] = acc;
}

// ---------------------------------------------------------------------------
// Kernel 2: fused  r = src@Wref^T -> u = tanh(q+r)·v -> logit = C*tanh(u)
// Grid: (S/64, B). Block: 256 threads = 8 waves.
// Block M-tile = 64 rows; wave = (msub, nchunk): 2 row-subtiles x 128 cols
// = 16 accumulator tiles; each B fragment feeds 2 WMMAs.
// Double-buffered LDS: TDM fills B[kk+1] and waves stage A[kk+1] while
// computing kk -> DMA and global loads hidden behind 16 WMMAs.
// ---------------------------------------------------------------------------
__global__ __launch_bounds__(256) void fused_scores_kernel(
    const float* __restrict__ src, const unsigned short* __restrict__ Wbf,
    const float* __restrict__ qbuf, const float* __restrict__ vvec,
    const unsigned char* __restrict__ maskin, const int* __restrict__ prev,
    float* __restrict__ out_logit, float* __restrict__ out_mask) {
  // Stride 40 elements = 80B rows: 16B-aligned fragment bases, and
  // (20*n mod 64) bank quads are perfectly disjoint -> conflict-free b128.
  __shared__ __attribute__((aligned(16))) unsigned short Ash[2][64][40];
  __shared__ __attribute__((aligned(16))) unsigned short Bsh[2][512][40];
  __shared__ float u_sh[64];

  const int b    = blockIdx.y;
  const int s0   = blockIdx.x * 64;
  const int tid  = threadIdx.x;
  const int lane = tid & 31;
  const int wv   = tid >> 5;
  const int msub = wv & 1;
  const int cbase = (wv >> 1) * 128;

  if (tid < 64) u_sh[tid] = 0.0f;           // covered by prologue barrier

  v8f acc0[8] = {};                         // rows msub*16 .. +15
  v8f acc1[8] = {};                         // rows 32+msub*16 .. +15

  const int col16  = lane & 15;
  const int khalfA = (lane >> 4) * 8;       // A frag: lanes 16-31 start at K=8
  const int khalfB = (lane >> 4) * 8;       // B frag: lanes 16-31 start at K=8
  const int arow0  = msub * 16 + col16;
  const int arow1  = arow0 + 32;

  const float* srcb = src + ((size_t)b * S_ + s0) * H_;

  // ---- constant parts of the TDM descriptor (D# group 1) ----
  const unsigned lds_b0 = (unsigned)(unsigned long long)(&Bsh[0][0][0]);
  const unsigned lds_b1 = (unsigned)(unsigned long long)(&Bsh[1][0][0]);
  const unsigned long long wbase = (unsigned long long)Wbf;
  i32x8 g1;
  g1[0] = (1 << 16)   // data_size = 2 bytes
        | (1 << 20)   // pad_enable
        | (3 << 22)   // pad_interval: 16 DWORDs (64B) between pads
        | (3 << 25);  // pad_amount:   4 DWORDs (16B)  -> 80B LDS row stride
  g1[1] = (H_ & 0xFFFF) << 16;   // tensor_dim0 = 512 (bits 79:48 lo half)
  g1[2] = (H_ & 0xFFFF) << 16;   // tensor_dim1 = 512 (bits 111:80 lo half)
  g1[3] = 32 << 16;              // tile_dim0 = 32 elements (64B rows)
  g1[4] = 512;                   // tile_dim1 = 512 rows
  g1[5] = H_;                    // tensor_dim0_stride = 512 elements
  g1[6] = 0;
  g1[7] = 0;

  const int row = tid >> 2;                 // A staging: 64 rows, 8 floats/thd
  const int c8  = (tid & 3) * 8;

  // ---- prologue: fill pipeline stage 0 ----
  if (wv == 0) tdm_issue_b(lds_b0, wbase, g1);
  {
    float4 a4 = *(const float4*)(srcb + (size_t)row * H_ + c8);
    float4 b4 = *(const float4*)(srcb + (size_t)row * H_ + c8 + 4);
    v4u p;
    p.x = (unsigned)f2bf(a4.x) | ((unsigned)f2bf(a4.y) << 16);
    p.y = (unsigned)f2bf(a4.z) | ((unsigned)f2bf(a4.w) << 16);
    p.z = (unsigned)f2bf(b4.x) | ((unsigned)f2bf(b4.y) << 16);
    p.w = (unsigned)f2bf(b4.z) | ((unsigned)f2bf(b4.w) << 16);
    *(v4u*)&Ash[0][row][c8] = p;
  }
  if (wv == 0) __builtin_amdgcn_s_wait_tensorcnt(0);
  __syncthreads();

  for (int kk = 0; kk < H_ / 32; ++kk) {
    const int cur = kk & 1, nxt = cur ^ 1;

    // ---- prefetch stage kk+1 (B via TDM, A via VALU convert) ----
    if (kk + 1 < H_ / 32) {
      if (wv == 0)
        tdm_issue_b(nxt ? lds_b1 : lds_b0,
                    wbase + (unsigned long long)(kk + 1) * 64ull, g1);
      const float* ap = srcb + (size_t)row * H_ + (kk + 1) * 32 + c8;
      float4 a4 = *(const float4*)(ap);
      float4 b4 = *(const float4*)(ap + 4);
      v4u p;
      p.x = (unsigned)f2bf(a4.x) | ((unsigned)f2bf(a4.y) << 16);
      p.y = (unsigned)f2bf(a4.z) | ((unsigned)f2bf(a4.w) << 16);
      p.z = (unsigned)f2bf(b4.x) | ((unsigned)f2bf(b4.y) << 16);
      p.w = (unsigned)f2bf(b4.z) | ((unsigned)f2bf(b4.w) << 16);
      *(v4u*)&Ash[nxt][row][c8] = p;
    }

    // ---- compute stage kk: 2 A frags, 8 B frags, 16 WMMAs ----
    const unsigned short (*Acur)[40] = Ash[cur];
    const unsigned short (*Bcur)[40] = Bsh[cur];

    v4u alo0 = *(const v4u*)&Acur[arow0][khalfA];
    v4u ahi0 = *(const v4u*)&Acur[arow0][khalfA + 16];
    v4u alo1 = *(const v4u*)&Acur[arow1][khalfA];
    v4u ahi1 = *(const v4u*)&Acur[arow1][khalfA + 16];
    v16bf af0 = __builtin_bit_cast(
        v16bf, __builtin_shufflevector(alo0, ahi0, 0, 1, 2, 3, 4, 5, 6, 7));
    v16bf af1 = __builtin_bit_cast(
        v16bf, __builtin_shufflevector(alo1, ahi1, 0, 1, 2, 3, 4, 5, 6, 7));

    // Hand-pipelined fragment loop: load ct+1 while WMMA-ing ct.
    v4u bl[2], bh[2];
    {
      int n0 = cbase + col16;
      bl[0] = *(const v4u*)&Bcur[n0][khalfB];
      bh[0] = *(const v4u*)&Bcur[n0][khalfB + 16];
    }
#pragma unroll
    for (int ct = 0; ct < 8; ++ct) {
      const int pc = ct & 1, pn = pc ^ 1;
      if (ct < 7) {
        int n = cbase + (ct + 1) * 16 + col16;
        bl[pn] = *(const v4u*)&Bcur[n][khalfB];
        bh[pn] = *(const v4u*)&Bcur[n][khalfB + 16];
      }
      v16bf bf = __builtin_bit_cast(
          v16bf, __builtin_shufflevector(bl[pc], bh[pc], 0, 1, 2, 3, 4, 5, 6, 7));
      acc0[ct] = __builtin_amdgcn_wmma_f32_16x16x32_bf16(
          false, af0, false, bf, (short)0, acc0[ct], false, false);
      acc1[ct] = __builtin_amdgcn_wmma_f32_16x16x32_bf16(
          false, af1, false, bf, (short)0, acc1[ct], false, false);
    }

    if (wv == 0 && kk + 1 < H_ / 32) __builtin_amdgcn_s_wait_tensorcnt(0);
    __syncthreads();                        // stage kk+1 ready, kk readers done
  }

  // ---- epilogue: u[s] += tanh(q[n] + r[s,n]) * v[n], reduce over n ----
  float part0[8] = {0, 0, 0, 0, 0, 0, 0, 0};
  float part1[8] = {0, 0, 0, 0, 0, 0, 0, 0};
  const int rhalf = (lane >> 4) * 8;        // C layout: lanes 16-31 = rows 8-15
#pragma unroll
  for (int ct = 0; ct < 8; ++ct) {
    int n = cbase + ct * 16 + col16;
    float qv = qbuf[b * H_ + n];
    float vv = vvec[n];
#pragma unroll
    for (int i = 0; i < 8; ++i) {
      part0[i] += fast_tanh(qv + acc0[ct][i]) * vv;
      part1[i] += fast_tanh(qv + acc1[ct][i]) * vv;
    }
  }
#pragma unroll
  for (int i = 0; i < 8; ++i) {
    atomicAdd(&u_sh[msub * 16 + rhalf + i], part0[i]);        // ds_add_f32
    atomicAdd(&u_sh[32 + msub * 16 + rhalf + i], part1[i]);
  }
  __syncthreads();

  if (tid < 64) {
    int s = s0 + tid;
    float lg = C_CLIP * fast_tanh(u_sh[tid]);
    bool m = (maskin[(size_t)b * S_ + s] != 0) | (prev[b] == s);
    size_t o = (size_t)b * S_ + s;
    out_logit[o] = m ? -__builtin_huge_valf() : lg;
    out_mask[o]  = m ? 1.0f : 0.0f;
  }
}

// ---------------------------------------------------------------------------
// Kernel 3: per-batch softmax over S, d_prime = probs @ src, tgt passthrough.
// ---------------------------------------------------------------------------
__global__ __launch_bounds__(256) void softmax_dprime_kernel(
    const float* __restrict__ src, const float* __restrict__ tgt,
    const float* __restrict__ logits, float* __restrict__ out_dprime,
    float* __restrict__ out_tgtT, float* __restrict__ out_probs) {
  __shared__ float p_sh[S_];
  __shared__ float red[256];
  const int b = blockIdx.x;
  const int tid = threadIdx.x;
  const float* lrow = logits + (size_t)b * S_;

  float mx = -__builtin_huge_valf();
  for (int i = tid; i < S_; i += 256) {
    float l = lrow[i];
    p_sh[i] = l;
    mx = fmaxf(mx, l);
  }
  red[tid] = mx;
  __syncthreads();
  for (int off = 128; off > 0; off >>= 1) {
    if (tid < off) red[tid] = fmaxf(red[tid], red[tid + off]);
    __syncthreads();
  }
  mx = red[0];
  __syncthreads();

  float sum = 0.0f;
  for (int i = tid; i < S_; i += 256) {
    float e = expf(p_sh[i] - mx);           // -inf -> 0
    p_sh[i] = e;
    sum += e;
  }
  red[tid] = sum;
  __syncthreads();
  for (int off = 128; off > 0; off >>= 1) {
    if (tid < off) red[tid] += red[tid + off];
    __syncthreads();
  }
  float inv = 1.0f / red[0];
  __syncthreads();

  for (int i = tid; i < S_; i += 256) {
    float p = p_sh[i] * inv;
    p_sh[i] = p;
    out_probs[(size_t)b * S_ + i] = p;
  }
  __syncthreads();

  float a0 = 0.0f, a1 = 0.0f;
  const float* sb = src + (size_t)b * S_ * H_;
  for (int s = 0; s < S_; ++s) {
    float p = p_sh[s];
    a0 += p * sb[(size_t)s * H_ + tid];
    a1 += p * sb[(size_t)s * H_ + tid + 256];
  }
  out_dprime[(size_t)b * H_ + tid]       = a0;
  out_dprime[(size_t)b * H_ + tid + 256] = a1;
  out_tgtT[(size_t)b * H_ + tid]         = tgt[(size_t)b * H_ + tid];
  out_tgtT[(size_t)b * H_ + tid + 256]   = tgt[(size_t)b * H_ + tid + 256];
}

// ---------------------------------------------------------------------------
extern "C" void kernel_launch(void* const* d_in, const int* in_sizes, int n_in,
                              void* d_out, int out_size, void* d_ws, size_t ws_size,
                              hipStream_t stream) {
  const float*         src  = (const float*)d_in[0];  // [B,S,H]
  const float*         tgt  = (const float*)d_in[1];  // [B,1,H]
  const unsigned char* mask = (const unsigned char*)d_in[2];  // [B,1,S] bool
  const int*           prev = (const int*)d_in[3];    // [B]
  const float*         Wref = (const float*)d_in[4];  // [H,H]
  const float*         Wq   = (const float*)d_in[5];  // [H,H]
  const float*         vvec = (const float*)d_in[6];  // [H]

  float* out        = (float*)d_out;
  float* out_dprime = out;                    // [1,B,H] == flat [B,H]
  float* out_tgtT   = out + B_ * H_;          // [1,B,H]
  float* out_probs  = out + 2 * B_ * H_;      // [B,1,S]
  float* out_logit  = out_probs + B_ * S_;    // [B,1,S]
  float* out_mask   = out_logit + B_ * S_;    // [B,1,S] as 0.0/1.0

  float*          qbuf = (float*)d_ws;                           // [B,H]
  unsigned short* Wbf  = (unsigned short*)((char*)d_ws + 65536); // [H,H] bf16

  wconv_kernel<<<dim3((H_ * H_) / 1024), dim3(256), 0, stream>>>(Wref, Wbf);
  q_kernel<<<dim3((B_ * H_) / 256), dim3(256), 0, stream>>>(tgt, Wq, qbuf);
  fused_scores_kernel<<<dim3(S_ / 64, B_), dim3(256), 0, stream>>>(
      src, Wbf, qbuf, vvec, mask, prev, out_logit, out_mask);
  softmax_dprime_kernel<<<dim3(B_), dim3(256), 0, stream>>>(
      src, tgt, out_logit, out_dprime, out_tgtT, out_probs);
}
